// HQNNModel_3307124818537
// MI455X (gfx1250) — matrix operand order repaired and checked by reading, not compiled, not run on hardware
//
#include <hip/hip_runtime.h>
#include <hip/hip_bf16.h>
#include <math.h>

// ---------------------------------------------------------------------------
// HQNN: MLP (WMMA f16 GEMMs, padded-f16 staging) -> 12-qubit LDS statevector
//   B=512, D=784 (pad 800), H=64, NQ=12 (pad 16), C=10, QAOA=1, Strong=4
// ---------------------------------------------------------------------------

typedef __attribute__((ext_vector_type(16))) _Float16 v16h;
typedef __attribute__((ext_vector_type(8)))  _Float16 v8h;
typedef __attribute__((ext_vector_type(8)))  float    v8f;

#define BATCH 512
#define DIN   784
#define DPAD  800
#define HID   64
#define NQ    12
#define NQP   16
#define NSTAT 4096
#define NCLS  10
#define LSTR  4
#define PI_F  3.14159265358979323846f

// ---------------------------------------------------------------------------
// Staging: f32 (rows x ksrc) -> f16 (rowspad x kdst), zero padded K and rows.
// ---------------------------------------------------------------------------
__global__ __launch_bounds__(256)
void pack_f16_pad(const float* __restrict__ src, _Float16* __restrict__ dst,
                  int rows, int rowspad, int ksrc, int kdst) {
    int i = blockIdx.x * 256 + threadIdx.x;
    if (i >= rowspad * kdst) return;
    int r = i / kdst, k = i - r * kdst;
    dst[i] = (r < rows && k < ksrc) ? (_Float16)src[r * ksrc + k]
                                    : (_Float16)0.0f;
}

// ---------------------------------------------------------------------------
// Fragment loaders (wave32, contiguous K order chosen here; A and B match):
//  A 16x32 f16, lane half h, row m=lane&15: e -> K=16h+e (one contiguous run)
//  B 32x16 f16, lane half h, col n=lane&15: e -> K=16h+e (one contiguous run)
// ---------------------------------------------------------------------------
__device__ inline v16h load_fragA(const _Float16* __restrict__ base) {
    v8h lo = *(const v8h*)(base);        // K = 16h .. 16h+7
    v8h hi = *(const v8h*)(base + 8);    // K = 16h+8 .. 16h+15
    return __builtin_shufflevector(lo, hi, 0,1,2,3,4,5,6,7,8,9,10,11,12,13,14,15);
}
__device__ inline v16h load_fragB(const _Float16* __restrict__ base) {
    v8h lo = *(const v8h*)(base);
    v8h hi = *(const v8h*)(base + 8);
    return __builtin_shufflevector(lo, hi, 0,1,2,3,4,5,6,7,8,9,10,11,12,13,14,15);
}

// ---------------------------------------------------------------------------
// Kernel 1: H1h = relu(X @ W1^T + b1) in f16.
// One wave per 16x64 row-tile: A fragment shared by 4 WMMAs per K-chunk.
// ---------------------------------------------------------------------------
__global__ __launch_bounds__(32)
void gemm1_relu(const _Float16* __restrict__ Xh,    // (512,800)
                const _Float16* __restrict__ W1h,   // (64,800)
                const float* __restrict__ bias,     // (64,)
                _Float16* __restrict__ H1h) {       // (512,64)
    const int lane = threadIdx.x & 31;
    const int half = lane >> 4;
    const int l15  = lane & 15;
    const int m0   = blockIdx.x * 16;

    const _Float16* arow = Xh + (m0 + l15) * DPAD + 16 * half;
    const _Float16* brow = W1h + l15 * DPAD + 16 * half;
    v8f c0 = {}, c1 = {}, c2 = {}, c3 = {};

    for (int k0 = 0; k0 < DPAD; k0 += 32) {
        v16h a  = load_fragA(arow + k0);
        v16h b0 = load_fragB(brow + k0);
        v16h b1 = load_fragB(brow + 16 * DPAD + k0);
        v16h b2 = load_fragB(brow + 32 * DPAD + k0);
        v16h b3 = load_fragB(brow + 48 * DPAD + k0);
        c0 = __builtin_amdgcn_wmma_f32_16x16x32_f16(false, a, false, b0, (short)0, c0, false, false);
        c1 = __builtin_amdgcn_wmma_f32_16x16x32_f16(false, a, false, b1, (short)0, c1, false, false);
        c2 = __builtin_amdgcn_wmma_f32_16x16x32_f16(false, a, false, b2, (short)0, c2, false, false);
        c3 = __builtin_amdgcn_wmma_f32_16x16x32_f16(false, a, false, b3, (short)0, c3, false, false);
    }

    v8f acc[4] = {c0, c1, c2, c3};
#pragma unroll
    for (int n = 0; n < 4; ++n) {
        int col = n * 16 + l15;
        float bv = bias[col];
#pragma unroll
        for (int r = 0; r < 8; ++r) {
            int m = m0 + half * 8 + r;
            float v = acc[n][r] + bv;
            H1h[m * HID + col] = (_Float16)(v > 0.0f ? v : 0.0f);
        }
    }
}

// ---------------------------------------------------------------------------
// Kernel 2: H2 = tanh(H1 @ W2^T + b2) * pi  (N=12 padded to 16, K=64)
// ---------------------------------------------------------------------------
__global__ __launch_bounds__(32)
void gemm2_tanh(const _Float16* __restrict__ H1h,   // (512,64)
                const _Float16* __restrict__ W2h,   // (16,64) rows 12..15 zero
                const float* __restrict__ b2,       // (12,)
                float* __restrict__ H2) {           // (512,12)
    const int lane = threadIdx.x & 31;
    const int half = lane >> 4;
    const int l15  = lane & 15;
    const int m0   = blockIdx.x * 16;

    const _Float16* arow = H1h + (m0 + l15) * HID + 16 * half;
    const _Float16* bcol = W2h + l15 * HID + 16 * half;
    v8f c = {};
#pragma unroll
    for (int k0 = 0; k0 < HID; k0 += 32) {
        v16h a = load_fragA(arow + k0);
        v16h b = load_fragB(bcol + k0);
        c = __builtin_amdgcn_wmma_f32_16x16x32_f16(false, a, false, b, (short)0, c, false, false);
    }
    if (l15 < NQ) {
        float bv = b2[l15];
#pragma unroll
        for (int r = 0; r < 8; ++r) {
            int m = m0 + half * 8 + r;
            H2[m * NQ + l15] = tanhf(c[r] + bv) * PI_F;
        }
    }
}

// ---------------------------------------------------------------------------
// Kernel 3: 12-qubit statevector sim, one workgroup (128 thr) per batch row;
// 32 KB state in LDS; final 12->10 projection fused. Wire q -> bit (11-q).
// ---------------------------------------------------------------------------
__device__ inline void apply_1q(float* sre, float* sim, int q,
                                float u00r, float u00i, float u01r, float u01i,
                                float u10r, float u10i, float u11r, float u11i,
                                int tid) {
    const int bitpos = (NQ - 1) - q;
    const int mask = 1 << bitpos;
    const int low  = mask - 1;
    for (int p = tid; p < (NSTAT >> 1); p += 128) {
        int i0 = ((p >> bitpos) << (bitpos + 1)) | (p & low);
        int i1 = i0 | mask;
        float r0 = sre[i0], m0 = sim[i0];
        float r1 = sre[i1], m1 = sim[i1];
        sre[i0] = u00r * r0 - u00i * m0 + u01r * r1 - u01i * m1;
        sim[i0] = u00r * m0 + u00i * r0 + u01r * m1 + u01i * r1;
        sre[i1] = u10r * r0 - u10i * m0 + u11r * r1 - u11i * m1;
        sim[i1] = u10r * m0 + u10i * r0 + u11r * m1 + u11i * r1;
    }
    __syncthreads();
}

__device__ inline void apply_zz(float* sre, float* sim, float theta,
                                int qi, int qj, int tid) {
    const int bi = (NQ - 1) - qi;
    const int bj = (NQ - 1) - qj;
    const float ch = __cosf(0.5f * theta);
    const float sh = __sinf(0.5f * theta);
    for (int s = tid; s < NSTAT; s += 128) {
        int diff = ((s >> bi) ^ (s >> bj)) & 1;
        float pi_ = diff ? sh : -sh;
        float r = sre[s], m = sim[s];
        sre[s] = ch * r - pi_ * m;
        sim[s] = ch * m + pi_ * r;
    }
    __syncthreads();
}

__device__ inline void apply_cnot(float* sre, float* sim, int qc, int qt, int tid) {
    const int mc = 1 << ((NQ - 1) - qc);
    const int mt = 1 << ((NQ - 1) - qt);
    for (int s = tid; s < NSTAT; s += 128) {
        if ((s & mc) && !(s & mt)) {
            int s2 = s | mt;
            float r = sre[s], m = sim[s];
            sre[s] = sre[s2]; sim[s] = sim[s2];
            sre[s2] = r;      sim[s2] = m;
        }
    }
    __syncthreads();
}

__global__ __launch_bounds__(128)
void quantum_layer(const float* __restrict__ H2,
                   const float* __restrict__ wq,
                   const float* __restrict__ wstr,
                   const float* __restrict__ Wp,
                   const float* __restrict__ bp,
                   float* __restrict__ out) {
    __shared__ float sre[NSTAT];
    __shared__ float simg[NSTAT];
    __shared__ float feats[NQ];
    __shared__ float swq[2 * NQ];
    __shared__ float sws[LSTR * NQ * 3];
    __shared__ float sev[NQ];

    const int tid = threadIdx.x;
    const int b   = blockIdx.x;

    for (int s = tid; s < NSTAT; s += 128) { sre[s] = 0.0f; simg[s] = 0.0f; }
    if (tid == 0) sre[0] = 1.0f;
    if (tid < NQ)            feats[tid] = H2[b * NQ + tid];
    if (tid < 2 * NQ)        swq[tid]   = wq[tid];
    if (tid < LSTR * NQ * 3) sws[tid]   = wstr[tid];
    if (tid < NQ)            sev[tid]   = 0.0f;
    __syncthreads();

    for (int q = 0; q < NQ; ++q) {                 // RX(feats)
        float t = feats[q];
        float c = __cosf(0.5f * t), s = __sinf(0.5f * t);
        apply_1q(sre, simg, q, c, 0.f, 0.f, -s, 0.f, -s, c, 0.f, tid);
    }
    for (int q = 0; q < NQ; ++q)                   // ZZ ring
        apply_zz(sre, simg, swq[q], q, (q + 1) % NQ, tid);
    for (int q = 0; q < NQ; ++q) {                 // RY(local)
        float t = swq[NQ + q];
        float c = __cosf(0.5f * t), s = __sinf(0.5f * t);
        apply_1q(sre, simg, q, c, 0.f, -s, 0.f, s, 0.f, c, 0.f, tid);
    }
    for (int q = 0; q < NQ; ++q) {                 // final RX
        float t = feats[q];
        float c = __cosf(0.5f * t), s = __sinf(0.5f * t);
        apply_1q(sre, simg, q, c, 0.f, 0.f, -s, 0.f, -s, c, 0.f, tid);
    }

    for (int l = 0; l < LSTR; ++l) {               // StronglyEntangling
        for (int q = 0; q < NQ; ++q) {
            const float* w = &sws[(l * NQ + q) * 3];
            float phi = w[0], th = w[1], omg = w[2];
            float c = __cosf(0.5f * th), s = __sinf(0.5f * th);
            float al = 0.5f * (phi + omg), be = 0.5f * (phi - omg);
            float ca = __cosf(al), sa = __sinf(al);
            float cb = __cosf(be), sb = __sinf(be);
            apply_1q(sre, simg, q,
                     ca * c, -sa * c,
                     -cb * s, -sb * s,
                     cb * s, -sb * s,
                     ca * c, sa * c, tid);
        }
        for (int q = 0; q < NQ; ++q)
            apply_cnot(sre, simg, q, (q + 1) % NQ, tid);
    }

    float acc[NQ];
#pragma unroll
    for (int q = 0; q < NQ; ++q) acc[q] = 0.0f;
    for (int s = tid; s < NSTAT; s += 128) {
        float p = sre[s] * sre[s] + simg[s] * simg[s];
#pragma unroll
        for (int q = 0; q < NQ; ++q)
            acc[q] += ((s >> ((NQ - 1) - q)) & 1) ? -p : p;
    }
#pragma unroll
    for (int q = 0; q < NQ; ++q) atomicAdd(&sev[q], acc[q]);
    __syncthreads();

    if (tid < NCLS) {
        float v = bp[tid];
#pragma unroll
        for (int q = 0; q < NQ; ++q) v += sev[q] * Wp[tid * NQ + q];
        out[b * NCLS + tid] = v;
    }
}

// ---------------------------------------------------------------------------
extern "C" void kernel_launch(void* const* d_in, const int* in_sizes, int n_in,
                              void* d_out, int out_size, void* d_ws, size_t ws_size,
                              hipStream_t stream) {
    (void)in_sizes; (void)n_in; (void)out_size; (void)ws_size;
    const float* x   = (const float*)d_in[0];
    const float* W1  = (const float*)d_in[1];
    const float* b1  = (const float*)d_in[2];
    const float* W2  = (const float*)d_in[3];
    const float* b2  = (const float*)d_in[4];
    const float* wq  = (const float*)d_in[5];
    const float* wst = (const float*)d_in[6];
    const float* Wp  = (const float*)d_in[7];
    const float* bp  = (const float*)d_in[8];
    float* out = (float*)d_out;

    char* ws = (char*)d_ws;
    _Float16* Xh  = (_Float16*)(ws);            // 512*800*2 = 819200 B
    _Float16* W1h = (_Float16*)(ws + 819200);   //  64*800*2 = 102400 B
    _Float16* W2h = (_Float16*)(ws + 921600);   //  16* 64*2 =   2048 B
    _Float16* H1h = (_Float16*)(ws + 923648);   // 512* 64*2 =  65536 B
    float*    H2  = (float*)   (ws + 989184);   // 512* 12*4 =  24576 B

    pack_f16_pad<<<(BATCH * DPAD + 255) / 256, 256, 0, stream>>>(x,  Xh,  BATCH, BATCH, DIN, DPAD);
    pack_f16_pad<<<(HID * DPAD + 255) / 256, 256, 0, stream>>>(W1, W1h, HID, HID, DIN, DPAD);
    pack_f16_pad<<<(NQP * HID + 255) / 256, 256, 0, stream>>>(W2, W2h, NQ, NQP, HID, HID);

    gemm1_relu<<<BATCH / 16, 32, 0, stream>>>(Xh, W1h, b1, H1h);
    gemm2_tanh<<<BATCH / 16, 32, 0, stream>>>(H1h, W2h, b2, H2);
    quantum_layer<<<BATCH, 128, 0, stream>>>(H2, wq, wst, Wp, bp, out);
}
